// Nonlocal_55851754717492
// MI455X (gfx1250) — matrix operand order repaired and checked by reading, not compiled, not run on hardware
//
#include <hip/hip_runtime.h>

// Non-local block, reassociated:  S = g·phiᵀ (128x128/batch),  W2 = w_rec·S/N,
// out = x + W2·theta + b_rec.
// Heavy GEMMs on V_WMMA_F32_16X16X32_F16 (f16 in, f32 accumulate);
// tiny W2 GEMM stays on V_WMMA_F32_16X16X4_F32.

#define BATCH  4
#define CCH    256      // input channels
#define CIN    128      // inter channels
#define NSP    4096     // spatial positions (64x64)
#define SPLITS 16       // split-K factor for the S GEMM

typedef __attribute__((ext_vector_type(2)))  float     v2f;
typedef __attribute__((ext_vector_type(8)))  float     v8f;
typedef __attribute__((ext_vector_type(8)))  _Float16  v8h;
typedef __attribute__((ext_vector_type(16))) _Float16  v16h;

// ---- WMMA helpers -----------------------------------------------------------
static __device__ __forceinline__ v8f wmma16(v16h a, v16h b, v8f c) {
  return __builtin_amdgcn_wmma_f32_16x16x32_f16(
      false, a, false, b, (short)0, c, false, false);
}
static __device__ __forceinline__ v8f wmma4(v2f a, v2f b, v8f c) {
  return __builtin_amdgcn_wmma_f32_16x16x4_f32(
      false, a, false, b, (short)0, c, false, false);
}
// concat two 8xf16 chunks (second at +off elements) into a v16h fragment
static __device__ __forceinline__ v16h cat16(const _Float16* p, int off) {
  v8h lo = *(const v8h*)p;
  v8h hi = *(const v8h*)(p + off);
  return __builtin_shufflevector(lo, hi, 0, 1, 2, 3, 4, 5, 6, 7,
                                 8, 9, 10, 11, 12, 13, 14, 15);
}
// Fragment lane mapping (wave32, ISA 7.12.2), f16 16x16x32:
//  A: lane l -> row M = l&15 ; K-runs [8h,8h+8) and [16+8h,16+8h+8), h=l>>4
//  B: lane l -> col N = l&15 ; K-runs [8h,8h+8) and [16+8h,16+8h+8)
//  C/D f32: reg r, lane l -> M = r + 8*(l>>4), N = l&15

// --- P0a: weights -> f16 row-major ------------------------------------------
__global__ void k_cvt_w(const float* __restrict__ a, const float* __restrict__ b,
                        const float* __restrict__ c,
                        _Float16* __restrict__ oa, _Float16* __restrict__ ob,
                        _Float16* __restrict__ oc) {
  const int i = blockIdx.x * 256 + threadIdx.x;   // 0 .. CIN*CCH-1
  oa[i] = (_Float16)a[i];
  ob[i] = (_Float16)b[i];
  oc[i] = (_Float16)c[i];
}

// --- P0b: xT_h[b][n][c] = (f16) x[b][c][n] ----------------------------------
// Register transpose: lane gathers one column via coalesced b32 reads, then
// writes a full 128B line of xT. grid (NSP/32, CCH/64, BATCH), block 32.
__global__ __launch_bounds__(32)
void k_xT(const float* __restrict__ x, _Float16* __restrict__ xTh) {
  const int l = threadIdx.x;
  const int n0 = blockIdx.x * 32, c0 = blockIdx.y * 64, b = blockIdx.z;
  const float* xb = x + (size_t)b * CCH * NSP;
  _Float16* ob = xTh + (size_t)b * NSP * CCH;
  v8h v[8];
#pragma unroll
  for (int jj = 0; jj < 8; ++jj)
#pragma unroll
    for (int q = 0; q < 8; ++q)
      v[jj][q] = (_Float16)xb[(size_t)(c0 + jj * 8 + q) * NSP + n0 + l];
  _Float16* op = ob + (size_t)(n0 + l) * CCH + c0;
#pragma unroll
  for (int jj = 0; jj < 8; ++jj) *(v8h*)(op + jj * 8) = v[jj];
}

// --- K1a: thetaT[b][n][o] = sum_c xT[n][c]*w_theta[o][c] + b_theta[o] --------
// Transposed orientation so K4's B operand is k-contiguous.
// grid (NSP/16, BATCH), block 32. Wave tile: 16(n) x 128(all o).
__global__ __launch_bounds__(32)
void k_thetaT(const _Float16* __restrict__ xTh, const _Float16* __restrict__ wh,
              const float* __restrict__ bias, _Float16* __restrict__ thT) {
  const int l = threadIdx.x, h = l >> 4, t = l & 15;
  const int n0 = blockIdx.x * 16, b = blockIdx.y;
  const _Float16* xb = xTh + (size_t)b * NSP * CCH;
  v8f acc[8] = {};
  for (int c0 = 0; c0 < CCH; c0 += 32) {
    const _Float16* ap = xb + (size_t)(n0 + t) * CCH + c0 + 8 * h;
    v16h a = cat16(ap, 16);
#pragma unroll
    for (int ot = 0; ot < 8; ++ot) {
      const _Float16* bp = wh + (size_t)(ot * 16 + t) * CCH + c0 + 8 * h;
      acc[ot] = wmma16(a, cat16(bp, 16), acc[ot]);
    }
  }
  _Float16* ob = thT + (size_t)b * NSP * CIN;
#pragma unroll
  for (int ot = 0; ot < 8; ++ot) {
    const float bv = bias[ot * 16 + t];
#pragma unroll
    for (int r = 0; r < 8; ++r)
      ob[(size_t)(n0 + r + 8 * h) * CIN + ot * 16 + t] =
          (_Float16)(acc[ot][r] + bv);
  }
}

// --- K1b: phi/g in natural layout p[b][o][n] (n contiguous = K2's K dim) -----
// grid (NSP/64, CIN/16, BATCH*2), block 32. Wave tile: 16(o) x 64(n).
__global__ __launch_bounds__(32)
void k_projn(const _Float16* __restrict__ xTh,
             const _Float16* __restrict__ wph, const float* __restrict__ bph,
             const _Float16* __restrict__ wgh, const float* __restrict__ bgh,
             _Float16* __restrict__ ph, _Float16* __restrict__ gh) {
  const int l = threadIdx.x, h = l >> 4, t = l & 15;
  const int n0 = blockIdx.x * 64, o0 = blockIdx.y * 16;
  const int b = blockIdx.z >> 1, which = blockIdx.z & 1;
  const _Float16* wh   = which ? wgh : wph;
  const float*    bias = which ? bgh : bph;
  _Float16* outp = (which ? gh : ph) + (size_t)b * CIN * NSP;
  const _Float16* xb = xTh + (size_t)b * NSP * CCH;
  v8f acc[4] = {};
  for (int c0 = 0; c0 < CCH; c0 += 32) {
    const _Float16* ap = wh + (size_t)(o0 + t) * CCH + c0 + 8 * h;
    v16h a = cat16(ap, 16);
#pragma unroll
    for (int sub = 0; sub < 4; ++sub) {
      const _Float16* bp = xb + (size_t)(n0 + sub * 16 + t) * CCH + c0 + 8 * h;
      acc[sub] = wmma16(a, cat16(bp, 16), acc[sub]);
    }
  }
#pragma unroll
  for (int r = 0; r < 8; ++r) {
    const float bv = bias[o0 + r + 8 * h];
    _Float16* dp = outp + (size_t)(o0 + r + 8 * h) * NSP + n0 + t;
#pragma unroll
    for (int sub = 0; sub < 4; ++sub)
      dp[sub * 16] = (_Float16)(acc[sub][r] + bv);
  }
}

// --- K2: split-K partials of S[b][c][k] = sum_m g[c][m]*phi[k][m] ------------
// grid (CIN/64, CIN/16, BATCH*SPLITS), block 32. Tile 16(c) x 64(k), K-chunk 256.
__global__ __launch_bounds__(32)
void k_spart(const _Float16* __restrict__ ph, const _Float16* __restrict__ gh,
             float* __restrict__ spart) {
  const int l = threadIdx.x, h = l >> 4, t = l & 15;
  const int k0 = blockIdx.x * 64, c0 = blockIdx.y * 16;
  const int b = blockIdx.z >> 4, split = blockIdx.z & (SPLITS - 1);
  const _Float16* pb = ph + (size_t)b * CIN * NSP;
  const _Float16* gb = gh + (size_t)b * CIN * NSP;
  const int m0 = split * (NSP / SPLITS);
  v8f acc[4] = {};
  for (int m = m0; m < m0 + NSP / SPLITS; m += 32) {
    const _Float16* ap = gb + (size_t)(c0 + t) * NSP + m + 8 * h;
    v16h a = cat16(ap, 16);
#pragma unroll
    for (int sub = 0; sub < 4; ++sub) {
      const _Float16* bp = pb + (size_t)(k0 + sub * 16 + t) * NSP + m + 8 * h;
      acc[sub] = wmma16(a, cat16(bp, 16), acc[sub]);
    }
  }
  float* sp = spart + (size_t)(b * SPLITS + split) * CIN * CIN;
#pragma unroll
  for (int r = 0; r < 8; ++r) {
    float* dp = sp + (size_t)(c0 + r + 8 * h) * CIN + k0 + t;
#pragma unroll
    for (int sub = 0; sub < 4; ++sub) dp[sub * 16] = acc[sub][r];
  }
}

// --- K2b: deterministic split-K reduction + 1/N scaling ----------------------
__global__ void k_reduce(const float* __restrict__ spart, float* __restrict__ S) {
  const int idx = blockIdx.x * blockDim.x + threadIdx.x;  // BATCH*CIN*CIN
  const int b = idx >> 14, rem = idx & 16383;             // CIN*CIN = 16384
  const float* sp = spart + (size_t)b * SPLITS * (CIN * CIN) + rem;
  float acc = 0.f;
#pragma unroll
  for (int s = 0; s < SPLITS; ++s) acc += sp[(size_t)s * (CIN * CIN)];
  S[idx] = acc * (1.0f / (float)NSP);
}

// --- K3: W2h[b][co][k] = (f16) sum_c w_rec[co][c]*S[b][c][k]  (f32 WMMA) -----
// grid (CIN/64, CCH/16, BATCH), block 32.
__global__ __launch_bounds__(32)
void k_w2(const float* __restrict__ w_rec, const float* __restrict__ S,
          _Float16* __restrict__ W2h) {
  const int l = threadIdx.x, h = l >> 4, t = l & 15;
  const int k0 = blockIdx.x * 64, co0 = blockIdx.y * 16, b = blockIdx.z;
  const float* Sb = S + (size_t)b * CIN * CIN;
  _Float16* W2b = W2h + (size_t)b * CCH * CIN;
  v8f acc[4] = {};
  for (int c0 = 0; c0 < CIN; c0 += 4) {
    const float* wr = w_rec + (size_t)(co0 + t) * CIN + c0 + 2 * h;
    v2f a; a.x = wr[0]; a.y = wr[1];
    const float* sp = Sb + (size_t)(c0 + 2 * h) * CIN + k0 + t;
    v2f b0, b1, b2, b3;
    b0.x = sp[0];  b0.y = sp[CIN];
    b1.x = sp[16]; b1.y = sp[CIN + 16];
    b2.x = sp[32]; b2.y = sp[CIN + 32];
    b3.x = sp[48]; b3.y = sp[CIN + 48];
    acc[0] = wmma4(a, b0, acc[0]); acc[1] = wmma4(a, b1, acc[1]);
    acc[2] = wmma4(a, b2, acc[2]); acc[3] = wmma4(a, b3, acc[3]);
  }
#pragma unroll
  for (int r = 0; r < 8; ++r) {
    _Float16* dp = W2b + (size_t)(co0 + r + 8 * h) * CIN + k0 + t;
#pragma unroll
    for (int sub = 0; sub < 4; ++sub) dp[sub * 16] = (_Float16)acc[sub][r];
  }
}

// --- K4: out[b][co][n] = x + b_rec[co] + sum_k W2h[co][k]*thetaT[n][k] -------
// grid (NSP/64, CCH/16, BATCH), block 32. Tile 16(co) x 64(n).
__global__ __launch_bounds__(32)
void k_final(const _Float16* __restrict__ W2h, const _Float16* __restrict__ thT,
             const float* __restrict__ x, const float* __restrict__ b_rec,
             float* __restrict__ out) {
  const int l = threadIdx.x, h = l >> 4, t = l & 15;
  const int n0 = blockIdx.x * 64, co0 = blockIdx.y * 16, b = blockIdx.z;
  const _Float16* W2b = W2h + (size_t)b * CCH * CIN;
  const _Float16* tb = thT + (size_t)b * NSP * CIN;
  const float* xb = x + (size_t)b * CCH * NSP;
  float* ob = out + (size_t)b * CCH * NSP;
  v8f acc[4] = {};
  for (int k0 = 0; k0 < CIN; k0 += 32) {
    const _Float16* ap = W2b + (size_t)(co0 + t) * CIN + k0 + 8 * h;
    v16h a = cat16(ap, 16);
#pragma unroll
    for (int sub = 0; sub < 4; ++sub) {
      const _Float16* bp = tb + (size_t)(n0 + sub * 16 + t) * CIN + k0 + 8 * h;
      acc[sub] = wmma16(a, cat16(bp, 16), acc[sub]);
    }
  }
#pragma unroll
  for (int r = 0; r < 8; ++r) {
    const int co = co0 + r + 8 * h;
    const float bv = b_rec[co];
    const float* xp = xb + (size_t)co * NSP + n0 + t;
    float* op = ob + (size_t)co * NSP + n0 + t;
#pragma unroll
    for (int sub = 0; sub < 4; ++sub)
      op[sub * 16] = acc[sub][r] + xp[sub * 16] + bv;
  }
}

extern "C" void kernel_launch(void* const* d_in, const int* in_sizes, int n_in,
                              void* d_out, int out_size, void* d_ws, size_t ws_size,
                              hipStream_t stream) {
  const float* x       = (const float*)d_in[0];
  const float* w_theta = (const float*)d_in[1];
  const float* b_theta = (const float*)d_in[2];
  const float* w_phi   = (const float*)d_in[3];
  const float* b_phi   = (const float*)d_in[4];
  const float* w_g     = (const float*)d_in[5];
  const float* b_g     = (const float*)d_in[6];
  const float* w_rec   = (const float*)d_in[7];
  const float* b_rec   = (const float*)d_in[8];
  float* out = (float*)d_out;

  // Workspace layout: ~24.7 MB total.
  _Float16* xTh = (_Float16*)d_ws;          // BATCH*NSP*CCH = 4,194,304 f16
  _Float16* thT = xTh + 4194304;            // BATCH*NSP*CIN = 2,097,152
  _Float16* ph  = thT + 2097152;            // BATCH*CIN*NSP = 2,097,152
  _Float16* gh  = ph  + 2097152;            // 2,097,152
  _Float16* wth = gh  + 2097152;            // CIN*CCH = 32,768
  _Float16* wph = wth + 32768;
  _Float16* wgh = wph + 32768;
  _Float16* W2h = wgh + 32768;              // BATCH*CCH*CIN = 131,072
  float* spart  = (float*)(W2h + 131072);   // SPLITS*BATCH*CIN*CIN = 1,048,576 f32
  float* S      = spart + 1048576;          // BATCH*CIN*CIN = 65,536 f32

  k_cvt_w <<<dim3((CIN * CCH) / 256), 256, 0, stream>>>(w_theta, w_phi, w_g,
                                                        wth, wph, wgh);
  k_xT    <<<dim3(NSP / 32, CCH / 64, BATCH), 32, 0, stream>>>(x, xTh);
  k_thetaT<<<dim3(NSP / 16, BATCH),          32, 0, stream>>>(xTh, wth, b_theta, thT);
  k_projn <<<dim3(NSP / 64, CIN / 16, BATCH * 2), 32, 0, stream>>>(xTh, wph, b_phi,
                                                                   wgh, b_g, ph, gh);
  k_spart <<<dim3(CIN / 64, CIN / 16, BATCH * SPLITS), 32, 0, stream>>>(ph, gh, spart);
  k_reduce<<<dim3((BATCH * CIN * CIN) / 256), 256, 0, stream>>>(spart, S);
  k_w2    <<<dim3(CIN / 64, CCH / 16, BATCH), 32, 0, stream>>>(w_rec, S, W2h);
  k_final <<<dim3(NSP / 64, CCH / 16, BATCH), 32, 0, stream>>>(W2h, thT, x, b_rec, out);
}
